// SparseMiddleExtractor_13134009991802
// MI455X (gfx1250) — matrix operand order repaired and checked by reading, not compiled, not run on hardware
//
#include <hip/hip_runtime.h>
#include <hip/hip_bf16.h>

// ---------------- problem constants ----------------
#define D1   11
#define HH   200
#define WW   176
#define HW_  (HH*WW)        // 35200
#define D2   5
#define D3   2
#define CIN0 128
#define CMID 64
#define NVOX 16000

typedef __attribute__((ext_vector_type(16))) __bf16 v16bf;
typedef __attribute__((ext_vector_type(8)))  float  v8f;

__device__ __forceinline__ unsigned short f2bf(float f) {
  unsigned u = __float_as_uint(f);
  unsigned r = u + 0x7FFFu + ((u >> 16) & 1u);   // round-to-nearest-even
  return (unsigned short)(r >> 16);
}

__device__ __forceinline__ void wait_async_le2() {
  asm volatile("s_wait_asynccnt 0x2" ::: "memory");
}
__device__ __forceinline__ void wait_async_le0() {
  asm volatile("s_wait_asynccnt 0x0" ::: "memory");
}

// ---------------- small utility kernels ----------------
__global__ void zero_i32(int* __restrict__ p, int n) {
  int i = blockIdx.x * 256 + threadIdx.x;
  if (i < n) p[i] = 0;
}

__global__ void scatter_grid(const int* __restrict__ coors, int* __restrict__ grid1) {
  int v = blockIdx.x * 256 + threadIdx.x;
  if (v >= NVOX) return;
  int z = coors[v*4+1], y = coors[v*4+2], x = coors[v*4+3];
  grid1[z*HW_ + y*WW + x] = v + 1;
}

__global__ void bn_prep(const float* __restrict__ g, const float* __restrict__ b,
                        const float* __restrict__ m, const float* __restrict__ v,
                        float* __restrict__ scale, float* __restrict__ shift) {
  int i = blockIdx.x * 64 + threadIdx.x;
  if (i >= 5*CMID) return;
  float s = g[i] * rsqrtf(v[i] + 1e-5f);
  scale[i] = s;
  shift[i] = b[i] - m[i]*s;
}

// Repack OIDHW weights into WMMA B-fragment layout:
// dst[((kc*4+nc)*32+lane)*16+j] = bf16(B[kc*32 + (lane/16)*16 + j][nc*16 + lane%16])
// with B[k][n] = w[(n*cin + k%cin)*ntaps + k/cin]   (K order: tap-major, channel-minor)
__global__ void repack_w(const float* __restrict__ w, unsigned short* __restrict__ dst,
                         int ntaps, int cin) {
  int e = blockIdx.x * 256 + threadIdx.x;
  int total = ntaps * cin * 64;
  if (e >= total) return;
  int j    = e & 15;
  int lane = (e >> 4) & 31;
  int nc   = (e >> 9) & 3;
  int kc   = e >> 11;
  int k = kc*32 + (lane >> 4)*16 + j;
  int n = nc*16 + (lane & 15);
  int c = k % cin, t = k / cin;
  dst[e] = f2bf(w[(n*cin + c)*ntaps + t]);
}

// Build compact active list for a stride-2-D, 3x1x1-VALID stage.
__global__ void build_list(const int* __restrict__ grid_in, int dout_,
                           int* __restrict__ grid_out, int* __restrict__ list,
                           int* __restrict__ counter) {
  int i = blockIdx.x * 256 + threadIdx.x;
  if (i >= dout_ * HW_) return;
  int d = i / HW_, rem = i % HW_;
  int base = (2*d)*HW_ + rem;
  int a = grid_in[base] | grid_in[base + HW_] | grid_in[base + 2*HW_];
  if (a) {
    int id = atomicAdd(counter, 1);
    list[id] = i;
    if (grid_out) grid_out[i] = id + 1;
  }
}

// ---------------- sparse implicit-GEMM conv with WMMA ----------------
// One wave computes a 16(M) x 64(N) tile, K = NTAPS*CIN in chunks of 32 (bf16).
// Software-pipelined: LDS A double-buffer + register B double-buffer.
// Single wave per workgroup => wave_barrier (scheduling-only fence) instead of
// s_barrier: DS ops execute in-order within a wave, and the async-to-LDS path
// is ordered explicitly with s_wait_asynccnt.
// MODE_NBR: 0 = conv1 (coords from coors, 3x3x3 pad1, grid1)
//           1 = 3x3x3 pad1 on stage-2 grid (rows from list, grid2)
//           2 = strided 3x1x1 (rows from output list, neighbors 2d+t in grid_in)
// MODE_IN : 0 = gather f32->bf16 via VGPRs; 1 = gather bf16 via async-to-LDS
// MODE_OUT: 0 = bf16 feat buffer; 1 = scatter f32 to d_out as channel n*2+d
template<int CIN, int NTAPS, int MODE_IN, int MODE_NBR, int MODE_OUT>
__global__ void __launch_bounds__(32)
conv_gemm(const float* __restrict__ vf,
          const unsigned short* __restrict__ featIn,
          const int* __restrict__ coors,
          const int* __restrict__ rows_list,
          const int* __restrict__ nrows_ptr, int nrows_const,
          const int* __restrict__ grid_nbr, int din_,
          const unsigned short* __restrict__ wfrag,
          const float* __restrict__ scale,
          const float* __restrict__ shift,
          unsigned short* __restrict__ featOut,
          float* __restrict__ dout)
{
  __shared__ int nbr[16 * NTAPS];
  __shared__ unsigned short At[2][16 * 40];  // 16 rows x 32 K, stride 40 u16:
                                             // 80B rows -> 16B aligned & conflict-free

  const int lane = threadIdx.x;
  const int nrows = nrows_ptr ? nrows_ptr[0] : nrows_const;
  const int row0 = blockIdx.x * 16;
  if (row0 >= nrows) return;

  // ---- precompute neighbor voxel ids for all 16 rows x NTAPS taps ----
  for (int idx = lane; idx < 16 * NTAPS; idx += 32) {
    int rr = idx / NTAPS, t = idx % NTAPS;
    int row = row0 + rr;
    int vid = -1;
    if (row < nrows) {
      if (MODE_NBR == 0) {
        int z = coors[row*4+1], y = coors[row*4+2], x = coors[row*4+3];
        int zz = z + t/9 - 1, yy = y + (t/3)%3 - 1, xx = x + t%3 - 1;
        if (zz >= 0 && zz < din_ && yy >= 0 && yy < HH && xx >= 0 && xx < WW)
          vid = grid_nbr[zz*HW_ + yy*WW + xx] - 1;
      } else if (MODE_NBR == 1) {
        int pos = rows_list[row];
        int d = pos / HW_, rem = pos % HW_, y = rem / WW, x = rem % WW;
        int zz = d + t/9 - 1, yy = y + (t/3)%3 - 1, xx = x + t%3 - 1;
        if (zz >= 0 && zz < din_ && yy >= 0 && yy < HH && xx >= 0 && xx < WW)
          vid = grid_nbr[zz*HW_ + yy*WW + xx] - 1;
      } else {
        int pos = rows_list[row];
        int d = pos / HW_, rem = pos % HW_;
        vid = grid_nbr[(2*d + t)*HW_ + rem] - 1;
      }
    }
    nbr[idx] = vid;
  }
  __builtin_amdgcn_wave_barrier();

  v8f acc[4] = {v8f{}, v8f{}, v8f{}, v8f{}};
  const int r = lane & 15;        // M row within tile / N column within n-chunk
  const int h = lane >> 4;        // K-half selector
  constexpr int KCH = (NTAPS * CIN) / 32;   // 108 / 54 / 6 : all even
  union BFrag { uint4 q[2]; v16bf v; };
  BFrag B0[4], B1[4];
  int vid0, vid1;                 // neighbor id pipelined alongside the buffers

  // Stage A slab for chunk kc into LDS buffer `buf` (unconditional; empty
  // neighbors use clamped address, zeros are substituted at fragment read).
  // Returns this lane's neighbor id for the chunk.
  auto stageA = [&](int kc, int buf) -> int {
    const int t  = (kc * 32) / CIN;
    const int cb = (kc * 32) % CIN;
    int vid = nbr[r * NTAPS + t];
    int svid = vid < 0 ? 0 : vid;
    unsigned short* dst = &At[buf][r * 40 + h * 16];
    if (MODE_IN == 0) {
      const float4* s = reinterpret_cast<const float4*>(vf + (size_t)svid*CIN0 + cb + h*16);
      unsigned* d32 = reinterpret_cast<unsigned*>(dst);
      #pragma unroll
      for (int q = 0; q < 4; ++q) {
        float4 f = s[q];
        d32[q*2+0] = (unsigned)f2bf(f.x) | ((unsigned)f2bf(f.y) << 16);
        d32[q*2+1] = (unsigned)f2bf(f.z) | ((unsigned)f2bf(f.w) << 16);
      }
    } else {
      // CDNA5 async copy global->LDS, no VGPR roundtrip; single offset applies
      // to both LDS and global address (ISA 08_async_tensor pseudocode).
      unsigned lds_off = (unsigned)(size_t)dst;
      unsigned long long ga =
          (unsigned long long)(size_t)(featIn + (size_t)svid*CMID + cb + h*16);
      asm volatile("global_load_async_to_lds_b128 %0, %1, off"
                   :: "v"(lds_off), "v"(ga) : "memory");
      asm volatile("global_load_async_to_lds_b128 %0, %1, off offset:16"
                   :: "v"(lds_off), "v"(ga) : "memory");
    }
    return vid;
  };

  auto loadB = [&](int kc, BFrag* Bf) {
    const uint4* bp = reinterpret_cast<const uint4*>(wfrag) + (size_t)kc * 256;
    #pragma unroll
    for (int nc = 0; nc < 4; ++nc) {
      const uint4* p2 = bp + nc*64 + lane*2;
      Bf[nc].q[0] = p2[0];
      Bf[nc].q[1] = p2[1];
    }
  };

  auto runWMMA = [&](int vcur, int buf, BFrag* Bf) {
    union AFrag { unsigned u[8]; v16bf v; } af;
    #pragma unroll
    for (int p = 0; p < 8; ++p) {
      // A layout (matches B repack): pair p -> K = half*16 + p*2
      int kb = (h << 4) + (p << 1);
      af.u[p] = *reinterpret_cast<const unsigned*>(&At[buf][r * 40 + kb]);
    }
    unsigned msk = (vcur >= 0) ? 0xFFFFFFFFu : 0u;
    #pragma unroll
    for (int p = 0; p < 8; ++p) af.u[p] &= msk;
    #pragma unroll
    for (int nc = 0; nc < 4; ++nc)
      acc[nc] = __builtin_amdgcn_wmma_f32_16x16x32_bf16(
          false, af.v, false, Bf[nc].v, (short)0, acc[nc], false, false);
  };

  // ---- pipelined main loop: 2 chunks per iteration ----
  vid0 = stageA(0, 0);
  loadB(0, B0);
  #pragma unroll 1
  for (int kc = 0; kc < KCH; kc += 2) {
    __builtin_amdgcn_wave_barrier();       // order: buf0 stage (kc) vs reads
    loadB(kc + 1, B1);
    vid1 = stageA(kc + 1, 1);
    if (MODE_IN == 1) wait_async_le2();    // buf0's asyncs done; buf1's in flight
    runWMMA(vid0, 0, B0);
    __builtin_amdgcn_wave_barrier();       // order: buf1 stage (kc+1) vs reads
    if (kc + 2 < KCH) {
      loadB(kc + 2, B0);
      vid0 = stageA(kc + 2, 0);
      if (MODE_IN == 1) wait_async_le2();  // buf1's asyncs done
    } else {
      if (MODE_IN == 1) wait_async_le0();
    }
    runWMMA(vid1, 1, B1);
  }

  // ---- fused BN + ReLU epilogue (C layout: VGPR e -> M=e+8*half, N=lane%16) ----
  #pragma unroll
  for (int nc = 0; nc < 4; ++nc) {
    const int n = nc * 16 + r;
    const float sc = scale[n], sh = shift[n];
    #pragma unroll
    for (int e = 0; e < 8; ++e) {
      int row = row0 + e + h * 8;
      if (row < nrows) {
        float val = acc[nc][e] * sc + sh;
        val = val > 0.f ? val : 0.f;
        if (MODE_OUT == 0) {
          featOut[(size_t)row * CMID + n] = f2bf(val);
        } else {
          int pos = rows_list[row];
          int d = pos / HW_, rem = pos % HW_;
          dout[(size_t)(n * D3 + d) * HW_ + rem] = val;   // reshape (c,d)->c*2+d
        }
      }
    }
  }
}

// ---------------- host-side orchestration ----------------
extern "C" void kernel_launch(void* const* d_in, const int* in_sizes, int n_in,
                              void* d_out, int out_size, void* d_ws, size_t ws_size,
                              hipStream_t stream) {
  const float* vf   = (const float*)d_in[0];
  const float* w1   = (const float*)d_in[1];
  const float* w2   = (const float*)d_in[2];
  const float* w3   = (const float*)d_in[3];
  const float* w4   = (const float*)d_in[4];
  const float* w5   = (const float*)d_in[5];
  const float* bn_g = (const float*)d_in[6];
  const float* bn_b = (const float*)d_in[7];
  const float* bn_m = (const float*)d_in[8];
  const float* bn_v = (const float*)d_in[9];
  const int*   coors = (const int*)d_in[10];
  float* out = (float*)d_out;

  size_t off = 0;
  auto alloc = [&](size_t bytes) -> void* {
    void* p = (char*)d_ws + off;
    off += (bytes + 255) & ~(size_t)255;
    return p;
  };
  int* grid1 = (int*)alloc((size_t)D1 * HW_ * 4);
  int* grid2 = (int*)alloc((size_t)D2 * HW_ * 4);
  int* list2 = (int*)alloc((size_t)NVOX * 4);
  int* list3 = (int*)alloc((size_t)NVOX * 4);
  int* counters = (int*)alloc(2 * 4);
  unsigned short* feat1 = (unsigned short*)alloc((size_t)NVOX * CMID * 2);
  unsigned short* feat2 = (unsigned short*)alloc((size_t)NVOX * CMID * 2);
  unsigned short* feat3 = (unsigned short*)alloc((size_t)NVOX * CMID * 2);
  unsigned short* feat4 = (unsigned short*)alloc((size_t)NVOX * CMID * 2);
  float* scales = (float*)alloc(5 * CMID * 4);
  float* shifts = (float*)alloc(5 * CMID * 4);
  unsigned short* w1p = (unsigned short*)alloc((size_t)27*CIN0*64 * 2);
  unsigned short* w2p = (unsigned short*)alloc((size_t)3 *CMID*64 * 2);
  unsigned short* w3p = (unsigned short*)alloc((size_t)27*CMID*64 * 2);
  unsigned short* w4p = (unsigned short*)alloc((size_t)27*CMID*64 * 2);
  unsigned short* w5p = (unsigned short*)alloc((size_t)3 *CMID*64 * 2);

  const int NOUT = 2 * CMID * HW_;   // 128*200*176 output floats

  // zero grids / counters / output
  zero_i32<<<(D1*HW_ + 255)/256, 256, 0, stream>>>(grid1, D1*HW_);
  zero_i32<<<(D2*HW_ + 255)/256, 256, 0, stream>>>(grid2, D2*HW_);
  zero_i32<<<1, 256, 0, stream>>>(counters, 2);
  zero_i32<<<(NOUT + 255)/256, 256, 0, stream>>>((int*)out, NOUT);

  scatter_grid<<<(NVOX + 255)/256, 256, 0, stream>>>(coors, grid1);
  bn_prep<<<(5*CMID + 63)/64, 64, 0, stream>>>(bn_g, bn_b, bn_m, bn_v, scales, shifts);

  repack_w<<<(27*CIN0*64 + 255)/256, 256, 0, stream>>>(w1, w1p, 27, CIN0);
  repack_w<<<(3 *CMID*64 + 255)/256, 256, 0, stream>>>(w2, w2p, 3,  CMID);
  repack_w<<<(27*CMID*64 + 255)/256, 256, 0, stream>>>(w3, w3p, 27, CMID);
  repack_w<<<(27*CMID*64 + 255)/256, 256, 0, stream>>>(w4, w4p, 27, CMID);
  repack_w<<<(3 *CMID*64 + 255)/256, 256, 0, stream>>>(w5, w5p, 3,  CMID);

  build_list<<<(D2*HW_ + 255)/256, 256, 0, stream>>>(grid1, D2, grid2, list2, counters + 0);

  const int NTILE = (NVOX + 15) / 16;   // 1000 tiles; later stages early-exit via counter

  // conv1: 128ch 3x3x3 pad1 at the 16000 voxel sites, BN0+ReLU -> feat1 (bf16)
  conv_gemm<CIN0, 27, 0, 0, 0><<<NTILE, 32, 0, stream>>>(
      vf, nullptr, coors, nullptr, nullptr, NVOX, grid1, D1,
      w1p, scales + 0, shifts + 0, feat1, nullptr);
  // conv2: 3x1x1 stride-2D VALID, BN1+ReLU -> feat2
  conv_gemm<CMID, 3, 1, 2, 0><<<NTILE, 32, 0, stream>>>(
      nullptr, feat1, nullptr, list2, counters + 0, 0, grid1, D1,
      w2p, scales + 64, shifts + 64, feat2, nullptr);
  // conv3: 3x3x3 pad1 on stage-2 grid, BN2+ReLU -> feat3
  conv_gemm<CMID, 27, 1, 1, 0><<<NTILE, 32, 0, stream>>>(
      nullptr, feat2, nullptr, list2, counters + 0, 0, grid2, D2,
      w3p, scales + 128, shifts + 128, feat3, nullptr);
  // conv4: 3x3x3 pad1, BN3+ReLU -> feat4
  conv_gemm<CMID, 27, 1, 1, 0><<<NTILE, 32, 0, stream>>>(
      nullptr, feat3, nullptr, list2, counters + 0, 0, grid2, D2,
      w4p, scales + 192, shifts + 192, feat4, nullptr);

  build_list<<<(D3*HW_ + 255)/256, 256, 0, stream>>>(grid2, D3, nullptr, list3, counters + 1);

  // conv5: 3x1x1 stride-2D VALID, BN4+ReLU, scatter f32 into (1,128,200,176) output
  conv_gemm<CMID, 3, 1, 2, 1><<<NTILE, 32, 0, stream>>>(
      nullptr, feat4, nullptr, list3, counters + 1, 0, grid2, D2,
      w5p, scales + 256, shifts + 256, nullptr, out);
}
